// LigerFusedLinearGRPOLoss_60215441490539
// MI455X (gfx1250) — matrix operand order repaired and checked by reading, not compiled, not run on hardware
//
#include <hip/hip_runtime.h>
#include <hip/hip_bf16.h>
#include <math.h>

// ---------------------------------------------------------------------------
// Problem constants: B=4, T=1024, H=2048, V=32000
// ---------------------------------------------------------------------------
#define B_DIM 4
#define T_DIM 1024
#define BT    4096
#define H_DIM 2048
#define V_DIM 32000

// Tiling
#define ROWS   64                 // M rows per workgroup
#define VTILE  1280               // N cols per workgroup
#define NVT    (V_DIM / VTILE)    // 25
#define NCH    256                // cols per chunk (8 waves x 32)
#define NCHUNK (VTILE / NCH)      // 5
#define KST    32                 // K per WMMA step
#define NWAVE  8
#define XS     40                 // LDS strides (ushorts), 16B aligned granules
#define WS     40

typedef __attribute__((ext_vector_type(16))) __bf16        v16bf;
typedef __attribute__((ext_vector_type(8)))  float         v8f;
typedef __attribute__((ext_vector_type(4)))  float         f4;
typedef __attribute__((ext_vector_type(8)))  unsigned short us8;
typedef __attribute__((ext_vector_type(16))) unsigned short us16;

__device__ __forceinline__ unsigned short f2bf(float f) {
  unsigned u = __builtin_bit_cast(unsigned, f);
  u += 0x7fffu + ((u >> 16) & 1u);
  return (unsigned short)(u >> 16);
}
__device__ __forceinline__ float bf2f(unsigned short h) {
  unsigned u = ((unsigned)h) << 16;
  return __builtin_bit_cast(float, u);
}

__device__ __forceinline__ v16bf ldfrag(const unsigned short* s, int o0, int o1) {
  us8 a = *(const us8*)(s + o0);
  us8 b = *(const us8*)(s + o1);
  us16 u;
#pragma unroll
  for (int i = 0; i < 8; ++i) { u[i] = a[i]; u[i + 8] = b[i]; }
  return __builtin_bit_cast(v16bf, u);
}

__device__ __forceinline__ v8f wmma_bf16(v16bf a, v16bf b, v8f c) {
  return __builtin_amdgcn_wmma_f32_16x16x32_bf16(false, a, false, b,
                                                 (short)0, c, false, false);
}

// ---------------------------------------------------------------------------
// Pack pass: fp32 -> (hi, lo) bf16 split, 8 elements / thread / iter
// ---------------------------------------------------------------------------
__global__ __launch_bounds__(256)
void pack_bf16(const float* __restrict__ src, unsigned short* __restrict__ hi,
               unsigned short* __restrict__ lo, long n8)
{
  long stride = (long)gridDim.x * blockDim.x;
  for (long i = (long)blockIdx.x * blockDim.x + threadIdx.x; i < n8; i += stride) {
    f4 a = ((const f4*)src)[i * 2 + 0];
    f4 b = ((const f4*)src)[i * 2 + 1];
    us8 h, l;
#pragma unroll
    for (int j = 0; j < 4; ++j) {
      unsigned short ha = f2bf(a[j]); h[j]     = ha; l[j]     = f2bf(a[j] - bf2f(ha));
      unsigned short hb = f2bf(b[j]); h[j + 4] = hb; l[j + 4] = f2bf(b[j] - bf2f(hb));
    }
    ((us8*)hi)[i] = h;
    ((us8*)lo)[i] = l;
  }
}

// ---------------------------------------------------------------------------
// Kernel 1: streaming logits (bf16x3 WMMA) + online-softmax partials
//   grid = (BT/ROWS, NVT), block = 256 (8 wave32)
//   PACKED: stage pre-split bf16 from workspace (pure b128 copies)
//   !PACKED: stage fp32 inputs with inline hi/lo split (fallback)
// ---------------------------------------------------------------------------
template <bool PACKED>
__global__ __launch_bounds__(256)
void grpo_main(const float* __restrict__ xf, const float* __restrict__ wf,
               const unsigned short* __restrict__ xh, const unsigned short* __restrict__ xl,
               const unsigned short* __restrict__ wh, const unsigned short* __restrict__ wl,
               const float* __restrict__ bias, const long long* __restrict__ input_id,
               float* __restrict__ pmax, float* __restrict__ psum,
               float* __restrict__ toklog)
{
  __shared__ unsigned short s_xhi[ROWS * XS];
  __shared__ unsigned short s_xlo[ROWS * XS];
  __shared__ unsigned short s_whi[NCH * WS];
  __shared__ unsigned short s_wlo[NCH * WS];
  __shared__ int   s_tid[ROWS];
  __shared__ float s_sm[NWAVE][ROWS];
  __shared__ float s_ss[NWAVE][ROWS];

  const int row0 = blockIdx.x * ROWS;
  const int v0   = blockIdx.y * VTILE;
  const int tid  = threadIdx.x;
  const int wave = tid >> 5;
  const int lane = tid & 31;
  const int half = lane >> 4;
  const int lr   = lane & 15;

  if (tid < ROWS) s_tid[tid] = (int)input_id[row0 + tid];

  float rmax[4][8], rsum[4][8];
#pragma unroll
  for (int m = 0; m < 4; ++m)
#pragma unroll
    for (int i = 0; i < 8; ++i) { rmax[m][i] = -3.0e38f; rsum[m][i] = 0.f; }

  for (int ch = 0; ch < NCHUNK; ++ch) {
    const int c0 = v0 + ch * NCH;

    v8f acc[4][2];
#pragma unroll
    for (int m = 0; m < 4; ++m)
#pragma unroll
      for (int t = 0; t < 2; ++t)
#pragma unroll
        for (int i = 0; i < 8; ++i) acc[m][t][i] = 0.f;

    for (int k0 = 0; k0 < H_DIM; k0 += KST) {
      __syncthreads();
      if (PACKED) {
        // ---- x tile: 64 rows x 4 granules (16B) per array; 1 granule/thread
        {
          int r = tid >> 2, q = (tid & 3) * 8;
          size_t g = (size_t)(row0 + r) * H_DIM + k0 + q;
          *(us8*)&s_xhi[r * XS + q] = *(const us8*)&xh[g];
          *(us8*)&s_xlo[r * XS + q] = *(const us8*)&xl[g];
        }
        // ---- W tile: 256 cols x 4 granules per array; 4 granules/thread
#pragma unroll
        for (int it = 0; it < 4; ++it) {
          int e = tid + it * 256;
          int c = e >> 2, q = (e & 3) * 8;
          size_t g = (size_t)(c0 + c) * H_DIM + k0 + q;
          *(us8*)&s_whi[c * WS + q] = *(const us8*)&wh[g];
          *(us8*)&s_wlo[c * WS + q] = *(const us8*)&wl[g];
        }
      } else {
#pragma unroll
        for (int it = 0; it < (ROWS * KST) / 256; ++it) {
          int e = tid + it * 256;
          int r = e >> 5, k = e & 31;
          float f = xf[(size_t)(row0 + r) * H_DIM + (k0 + k)];
          unsigned short hv = f2bf(f);
          s_xhi[r * XS + k] = hv;
          s_xlo[r * XS + k] = f2bf(f - bf2f(hv));
        }
#pragma unroll
        for (int it = 0; it < (NCH * KST) / 256; ++it) {
          int e = tid + it * 256;
          int c = e >> 5, k = e & 31;
          float f = wf[(size_t)(c0 + c) * H_DIM + (k0 + k)];
          unsigned short hv = f2bf(f);
          s_whi[c * WS + k] = hv;
          s_wlo[c * WS + k] = f2bf(f - bf2f(hv));
        }
      }
      __syncthreads();

      // B fragments: two 16-col tiles per wave
      v16bf bhi[2], blo[2];
#pragma unroll
      for (int t = 0; t < 2; ++t) {
        const int cb = (wave * 32 + t * 16 + lr) * WS + half * 16;
        bhi[t] = ldfrag(s_whi, cb, cb + 8);
        blo[t] = ldfrag(s_wlo, cb, cb + 8);
      }
#pragma unroll
      for (int m = 0; m < 4; ++m) {
        const int ab = (m * 16 + lr) * XS + half * 16;
        v16bf ahi = ldfrag(s_xhi, ab, ab + 8);
        v16bf alo = ldfrag(s_xlo, ab, ab + 8);
#pragma unroll
        for (int t = 0; t < 2; ++t) {
          acc[m][t] = wmma_bf16(ahi, bhi[t], acc[m][t]);   // hi*hi
          acc[m][t] = wmma_bf16(ahi, blo[t], acc[m][t]);   // hi*lo
          acc[m][t] = wmma_bf16(alo, bhi[t], acc[m][t]);   // lo*hi
        }
      }
    }

    // ---- per-chunk row stats (32 cols per wave) + token-logit capture ----
    const int   colg0 = c0 + wave * 32 + lr;
    const int   colg1 = colg0 + 16;
    const float b0 = bias[colg0];
    const float b1 = bias[colg1];
#pragma unroll
    for (int m = 0; m < 4; ++m) {
#pragma unroll
      for (int i = 0; i < 8; ++i) {
        const int rowl = m * 16 + half * 8 + i;
        float v0v = acc[m][0][i] + b0;
        float v1v = acc[m][1][i] + b1;
        const int tk = s_tid[rowl];
        if (tk == colg0) toklog[row0 + rowl] = v0v;
        if (tk == colg1) toklog[row0 + rowl] = v1v;
        float cm = fmaxf(v0v, v1v);
#pragma unroll
        for (int s = 1; s < 16; s <<= 1) cm = fmaxf(cm, __shfl_xor(cm, s));
        float es = __expf(v0v - cm) + __expf(v1v - cm);
#pragma unroll
        for (int s = 1; s < 16; s <<= 1) es += __shfl_xor(es, s);
        float nm = fmaxf(rmax[m][i], cm);
        rsum[m][i] = rsum[m][i] * __expf(rmax[m][i] - nm) + es * __expf(cm - nm);
        rmax[m][i] = nm;
      }
    }
  }

  // ---- cross-wave merge, write V-tile partials ----
  if (lr == 0) {
#pragma unroll
    for (int m = 0; m < 4; ++m)
#pragma unroll
      for (int i = 0; i < 8; ++i) {
        int r = m * 16 + half * 8 + i;
        s_sm[wave][r] = rmax[m][i];
        s_ss[wave][r] = rsum[m][i];
      }
  }
  __syncthreads();
  if (tid < ROWS) {
    float M = -3.0e38f, S = 0.f;
#pragma unroll
    for (int wv = 0; wv < NWAVE; ++wv) {
      float m_ = s_sm[wv][tid], s_ = s_ss[wv][tid];
      float nm = fmaxf(M, m_);
      S = S * __expf(M - nm) + s_ * __expf(m_ - nm);
      M = nm;
    }
    pmax[(size_t)blockIdx.y * BT + row0 + tid] = M;
    psum[(size_t)blockIdx.y * BT + row0 + tid] = S;
  }
}

// ---------------------------------------------------------------------------
// Kernel 2: per-row lse merge + GRPO loss + block partial sums
// ---------------------------------------------------------------------------
__global__ __launch_bounds__(256)
void grpo_finalize(const float* __restrict__ pmax, const float* __restrict__ psum,
                   const float* __restrict__ toklog, const int* __restrict__ attn,
                   const float* __restrict__ adv, const float* __restrict__ refl,
                   const float* __restrict__ oldl, float* __restrict__ out,
                   float* __restrict__ blockred)
{
  __shared__ float s_red[NWAVE][4];
  const int rowg = blockIdx.x * 256 + threadIdx.x;

  float M = -3.0e38f, S = 0.f;
#pragma unroll
  for (int vt = 0; vt < NVT; ++vt) {
    float m_ = pmax[(size_t)vt * BT + rowg];
    float s_ = psum[(size_t)vt * BT + rowg];
    float nm = fmaxf(M, m_);
    S = S * __expf(M - nm) + s_ * __expf(m_ - nm);
    M = nm;
  }
  const float lse = M + __logf(S);
  const float ptl = toklog[rowg] - lse;
  out[1 + rowg] = ptl;

  const float mask = (float)attn[rowg];
  const float a    = adv[rowg / T_DIM];
  const float d    = refl[rowg] - ptl;
  const float kl   = __expf(d) - d - 1.f;
  const float c1   = __expf(ptl - oldl[rowg]);
  const float c2   = fminf(fmaxf(c1, 0.8f), 1.2f);
  const float l1   = c1 * a, l2 = c2 * a;
  const float pls  = -(fminf(l1, l2) - 0.04f * kl);
  const float clip = (l1 < l2) ? 1.f : 0.f;

  float v0 = pls * mask, v1 = kl * mask, v2 = clip * mask, v3 = mask;
#pragma unroll
  for (int s = 1; s < 32; s <<= 1) {
    v0 += __shfl_xor(v0, s); v1 += __shfl_xor(v1, s);
    v2 += __shfl_xor(v2, s); v3 += __shfl_xor(v3, s);
  }
  const int wave = threadIdx.x >> 5, lane = threadIdx.x & 31;
  if (lane == 0) { s_red[wave][0] = v0; s_red[wave][1] = v1;
                   s_red[wave][2] = v2; s_red[wave][3] = v3; }
  __syncthreads();
  if (threadIdx.x == 0) {
    float t0 = 0, t1 = 0, t2 = 0, t3 = 0;
#pragma unroll
    for (int wv = 0; wv < NWAVE; ++wv) {
      t0 += s_red[wv][0]; t1 += s_red[wv][1];
      t2 += s_red[wv][2]; t3 += s_red[wv][3];
    }
    blockred[blockIdx.x * 4 + 0] = t0;
    blockred[blockIdx.x * 4 + 1] = t1;
    blockred[blockIdx.x * 4 + 2] = t2;
    blockred[blockIdx.x * 4 + 3] = t3;
  }
}

// ---------------------------------------------------------------------------
// Kernel 3: final scalars
// ---------------------------------------------------------------------------
__global__ void grpo_scalars(const float* __restrict__ blockred,
                             float* __restrict__ out)
{
  if (threadIdx.x == 0) {
    float l = 0, k = 0, c = 0, ms = 0;
    for (int b = 0; b < BT / 256; ++b) {
      l  += blockred[b * 4 + 0];
      k  += blockred[b * 4 + 1];
      c  += blockred[b * 4 + 2];
      ms += blockred[b * 4 + 3];
    }
    out[0]      = l / ms;
    out[1 + BT] = k / ms;
    out[2 + BT] = c / ms;
  }
}

// ---------------------------------------------------------------------------
extern "C" void kernel_launch(void* const* d_in, const int* in_sizes, int n_in,
                              void* d_out, int out_size, void* d_ws, size_t ws_size,
                              hipStream_t stream)
{
  const float*     x        = (const float*)d_in[0];
  const float*     w        = (const float*)d_in[1];
  const float*     bias     = (const float*)d_in[2];
  const long long* input_id = (const long long*)d_in[3];
  const int*       attn     = (const int*)d_in[4];
  const float*     adv      = (const float*)d_in[5];
  const float*     refl     = (const float*)d_in[6];
  const float*     oldl     = (const float*)d_in[7];
  float*           out      = (float*)d_out;

  // Workspace layout
  float* pmax = (float*)d_ws;                       // NVT*BT
  float* psum = pmax + (size_t)NVT * BT;            // NVT*BT
  float* tok  = psum + (size_t)NVT * BT;            // BT
  float* bred = tok + BT;                           // 16*4
  size_t base_floats = (size_t)NVT * BT * 2 + BT + 64;
  // align packed region to 256B
  size_t off = (base_floats * 4 + 255) & ~(size_t)255;
  unsigned short* xh = (unsigned short*)((char*)d_ws + off);
  unsigned short* xl = xh + (size_t)BT * H_DIM;
  unsigned short* wh = xl + (size_t)BT * H_DIM;
  unsigned short* wl = wh + (size_t)V_DIM * H_DIM;
  size_t need = off + ((size_t)BT * H_DIM * 2 + (size_t)V_DIM * H_DIM * 2) * sizeof(unsigned short);
  const bool packed = ws_size >= need;

  dim3 grid(BT / ROWS, NVT);
  if (packed) {
    pack_bf16<<<1024, 256, 0, stream>>>(x, xh, xl, (long)BT * H_DIM / 8);
    pack_bf16<<<4096, 256, 0, stream>>>(w, wh, wl, (long)V_DIM * H_DIM / 8);
    grpo_main<true><<<grid, 256, 0, stream>>>(x, w, xh, xl, wh, wl, bias, input_id,
                                              pmax, psum, tok);
  } else {
    grpo_main<false><<<grid, 256, 0, stream>>>(x, w, xh, xl, wh, wl, bias, input_id,
                                               pmax, psum, tok);
  }
  grpo_finalize<<<BT / 256, 256, 0, stream>>>(pmax, psum, tok, attn, adv, refl, oldl,
                                              out, bred);
  grpo_scalars<<<1, 32, 0, stream>>>(bred, out);
}
